// _MultiHeadAttention_4105988735944
// MI455X (gfx1250) — compile-verified, matrix operand order checked by reading, not run on hardware
//
#include <hip/hip_runtime.h>
#include <hip/hip_bf16.h>

// ---------------------------------------------------------------------------
// MHA layer for MI455X (gfx1250, wave32, WMMA f32<-f16 16x16x32).
// Pipeline: [wtrans] -> [qkv+rope] -> [scores+softmax+attnV] -> [oproj]
// GEMM kernels use 16x64 per-wave strips (4 accumulators) so each A-fragment
// load/convert is amortized over 4 WMMAs.
// Workspace layout (f16): 4x WT (1M elems each) + qf16/kf16/vT/o (4M each)
//   = 20,971,520 f16 = 40 MB.
// ---------------------------------------------------------------------------

#define BSZ 2
#define SEQ 2048
#define DIM 1024
#define NH  16
#define DK  64
#define SCALE 0.125f            // DK^-0.5
#define LN_ROPE_BASE 9.210340371976184f  // ln(10000)

typedef __attribute__((ext_vector_type(16))) _Float16 v16h;
typedef __attribute__((ext_vector_type(8)))  float    v8f;

__device__ __forceinline__ v8f wmma_f16(v16h a, v16h b, v8f c) {
  return __builtin_amdgcn_wmma_f32_16x16x32_f16(
      /*neg_a=*/false, a, /*neg_b=*/false, b,
      /*c_mod=*/(short)0, c, /*reuse_a=*/false, /*reuse_b=*/false);
}

// A fragment (16x32, row-major, stride ld in f16 elems) per ISA 7.12.2:
// lanes 0-15 row=lane, K={0..7,16..23}; lanes 16-31 row=lane-16, K={8..15,24..31}
__device__ __forceinline__ v16h frag_f16(const _Float16* __restrict__ base,
                                         int ld, int lane) {
  int r  = lane & 15;
  int k0 = (lane & 16) ? 8 : 0;
  const _Float16* p = base + r * ld + k0;
  v16h f;
#pragma unroll
  for (int i = 0; i < 8; ++i) f[i] = p[i];
#pragma unroll
  for (int i = 0; i < 8; ++i) f[i + 8] = p[16 + i];
  return f;
}

// Same fragment but sourced from f32 (global or LDS), converted per element.
__device__ __forceinline__ v16h frag_f32(const float* __restrict__ base,
                                         int ld, int lane) {
  int r  = lane & 15;
  int k0 = (lane & 16) ? 8 : 0;
  const float* p = base + r * ld + k0;
  v16h f;
#pragma unroll
  for (int i = 0; i < 8; ++i) f[i] = (_Float16)p[i];
#pragma unroll
  for (int i = 0; i < 8; ++i) f[i + 8] = (_Float16)p[16 + i];
  return f;
}

// ---------------------------------------------------------------------------
// Kernel 0: W[k][n] (f32) -> WT[n][k] (f16) for Wq, Wk, Wv, Wo.
// ---------------------------------------------------------------------------
__global__ void wtrans_kernel(const float* __restrict__ Wq,
                              const float* __restrict__ Wk,
                              const float* __restrict__ Wv,
                              const float* __restrict__ Wo,
                              _Float16* __restrict__ WqT,
                              _Float16* __restrict__ WkT,
                              _Float16* __restrict__ WvT,
                              _Float16* __restrict__ WoT) {
  const float* src; _Float16* dst;
  switch (blockIdx.y) {
    case 0:  src = Wq; dst = WqT; break;
    case 1:  src = Wk; dst = WkT; break;
    case 2:  src = Wv; dst = WvT; break;
    default: src = Wo; dst = WoT; break;
  }
  int idx = blockIdx.x * blockDim.x + threadIdx.x;   // 0 .. 1024*1024
  int n = idx >> 10;
  int k = idx & 1023;
  dst[n * DIM + k] = (_Float16)src[k * DIM + n];     // coalesced writes
}

// ---------------------------------------------------------------------------
// Kernel 1: QKV projection + bias + RoPE(q,k).
//   mode(blockIdx.z): 0 -> q (rope) [b,h,s,dk]   f16
//                     1 -> k (rope) [b,h,s,dk]   f16
//                     2 -> v        [b,h,dk,s]   f16 (transposed for attn*V B)
// One wave per 16x64 output strip (4 WMMA accumulators); K=1024 in steps of 32.
// A fragment (f32->f16 convert of X) is loaded once and reused by 4 WMMAs.
// ---------------------------------------------------------------------------
__global__ void qkv_rope_kernel(const float* __restrict__ X,       // [4096,1024]
                                const _Float16* __restrict__ WqT,
                                const _Float16* __restrict__ WkT,
                                const _Float16* __restrict__ WvT,
                                const float* __restrict__ bq,
                                const float* __restrict__ bk,
                                const float* __restrict__ bv,
                                _Float16* __restrict__ qf16,
                                _Float16* __restrict__ kf16,
                                _Float16* __restrict__ vT) {
  const int lane  = threadIdx.x & 31;
  const int wave  = threadIdx.x >> 5;
  const int mode  = blockIdx.z;
  const int m0    = blockIdx.x * 16;
  const int strip = blockIdx.y * 4 + wave;    // 0..15
  const int n0    = strip * 64;

  const _Float16* WT; const float* bias;
  if (mode == 0)      { WT = WqT; bias = bq; }
  else if (mode == 1) { WT = WkT; bias = bk; }
  else                { WT = WvT; bias = bv; }

  v8f acc[4] = {};
#pragma unroll 2
  for (int k = 0; k < DIM; k += 32) {
    v16h a = frag_f32(X + m0 * DIM + k, DIM, lane);   // X tile, f32->f16 once
#pragma unroll
    for (int j = 0; j < 4; ++j) {
      v16h b = frag_f16(WT + (n0 + 16 * j) * DIM + k, DIM, lane);
      acc[j] = wmma_f16(a, b, acc[j]);
    }
  }

  const int mb = (lane & 16) ? 8 : 0;
#pragma unroll
  for (int j = 0; j < 4; ++j) {
    const int n  = n0 + 16 * j + (lane & 15);
    const int h  = n / DK;
    const int d  = n % DK;
    const float bs = bias[n];
    // RoPE frequency for this column's pair: 10000^(-(2p)/DK), 2p = d&~1
    const float freq = __expf(-(float)(d & ~1) * (LN_ROPE_BASE / (float)DK));
#pragma unroll
    for (int i = 0; i < 8; ++i) {
      int r  = m0 + mb + i;          // flat row = b*SEQ + s
      int b_ = r >> 11;
      int s  = r & (SEQ - 1);
      float val   = acc[j][i] + bs;
      float other = __shfl_xor(val, 1, 32);  // paired column d^1
      if (mode < 2) {
        float c, sn;
        __sincosf((float)s * freq, &sn, &c);
        float o = (d & 1) ? (val * c + other * sn)   // x1*cos + x0*sin
                          : (val * c - other * sn);  // x0*cos - x1*sin
        _Float16* dst = (mode == 0) ? qf16 : kf16;
        dst[(((b_ * NH + h) * SEQ + s) * DK) + d] = (_Float16)o;
      } else {
        vT[(((b_ * NH + h) * DK + d) * (size_t)SEQ) + s] = (_Float16)val;
      }
    }
  }
}

// ---------------------------------------------------------------------------
// Kernel 2: per (b, h, 16-row q-tile):
//   scores = q*k^T*scale + prev  -> LDS (16x2048 f32) and d_out scores region
//   softmax rows in LDS (shuffle reductions, 16 lanes per row)
//   o_tile = probs @ V via WMMA (K=2048 split across wave pairs, LDS reduce)
// 256 threads = 8 waves; dynamic LDS = 16*2048*4 + 8*256*4 = 139264 B.
// ---------------------------------------------------------------------------
__global__ void attn_kernel(const _Float16* __restrict__ qf16,  // [b,h,s,dk]
                            const _Float16* __restrict__ kf16,  // [b,h,s,dk]
                            const _Float16* __restrict__ vT,    // [b,h,dk,s]
                            const float* __restrict__ prev,     // [b,h,s,s]
                            float* __restrict__ scores_out,     // [b,h,s,s]
                            _Float16* __restrict__ o) {         // [b*s, h*dk]
  extern __shared__ float smem[];
  float* sc  = smem;                 // [16][2048]
  float* red = smem + 16 * SEQ;      // [4][16][16] partial C tiles

  const int lane = threadIdx.x & 31;
  const int wave = threadIdx.x >> 5;
  const int qt = blockIdx.x, h = blockIdx.y, b = blockIdx.z;
  const int q0 = qt * 16;
  const int bh = b * NH + h;

  const _Float16* qbase = qf16 + ((size_t)(bh * SEQ + q0)) * DK;  // [16][64]
  const _Float16* kbase = kf16 + ((size_t)bh * SEQ) * DK;         // [2048][64]
  const float*    pbase = prev + ((size_t)bh * SEQ + q0) * SEQ;   // [16][2048]
  float*          sobase = scores_out + ((size_t)bh * SEQ + q0) * SEQ;

  // ---- Phase 1: scores (128 N-tiles of 16, round-robin over 8 waves) ----
  v16h a0 = frag_f16(qbase + 0, DK, lane);    // K = 0..31 of dk
  v16h a1 = frag_f16(qbase + 32, DK, lane);   // K = 32..63
  const int nlane = lane & 15;
  const int mb    = (lane & 16) ? 8 : 0;
  for (int t = wave; t < SEQ / 16; t += 8) {
    int n0 = t * 16;
    v16h b0 = frag_f16(kbase + n0 * DK + 0, DK, lane);
    v16h b1 = frag_f16(kbase + n0 * DK + 32, DK, lane);
    v8f c = {};
    c = wmma_f16(a0, b0, c);
    c = wmma_f16(a1, b1, c);
    int n = n0 + nlane;
#pragma unroll
    for (int i = 0; i < 8; ++i) {
      int r = mb + i;
      float v = c[i] * SCALE + pbase[(size_t)r * SEQ + n];
      sc[r * SEQ + n] = v;
      sobase[(size_t)r * SEQ + n] = v;
    }
  }
  __syncthreads();

  // ---- Phase 2: softmax over each of 16 rows (16 lanes per row) ----
  {
    int row = threadIdx.x >> 4;   // 0..15
    int sub = threadIdx.x & 15;
    float* rp = sc + row * SEQ;
    float mx = -3.0e38f;
    for (int j = sub; j < SEQ; j += 16) mx = fmaxf(mx, rp[j]);
#pragma unroll
    for (int m = 8; m >= 1; m >>= 1) mx = fmaxf(mx, __shfl_xor(mx, m, 32));
    float sum = 0.0f;
    for (int j = sub; j < SEQ; j += 16) {
      float e = __expf(rp[j] - mx);
      rp[j] = e;
      sum += e;
    }
#pragma unroll
    for (int m = 8; m >= 1; m >>= 1) sum += __shfl_xor(sum, m, 32);
    float inv = 1.0f / sum;
    for (int j = sub; j < SEQ; j += 16) rp[j] *= inv;
  }
  __syncthreads();

  // ---- Phase 3: o = probs @ V. waves 0-3: dk tiles, K=0..1023;
  //                              waves 4-7: same tiles, K=1024..2047 ----
  const int nt = wave & 3;          // which 16-wide dk tile
  const int kh = wave >> 2;         // which K half
  const _Float16* vbase = vT + ((size_t)(bh * DK + nt * 16)) * SEQ;
  v8f c = {};
  for (int k = kh * (SEQ / 2); k < (kh + 1) * (SEQ / 2); k += 32) {
    v16h a = frag_f32(sc + k, SEQ, lane);       // probs (f32 LDS -> f16)
    v16h bb = frag_f16(vbase + k, SEQ, lane);   // V^T rows = dk columns
    c = wmma_f16(a, bb, c);
  }
  if (wave >= 4) {
#pragma unroll
    for (int i = 0; i < 8; ++i)
      red[(wave - 4) * 256 + (mb + i) * 16 + nlane] = c[i];
  }
  __syncthreads();
  if (wave < 4) {
#pragma unroll
    for (int i = 0; i < 8; ++i) {
      int r = mb + i;
      float val = c[i] + red[wave * 256 + r * 16 + nlane];
      // o layout: [(b*SEQ + s) * 1024 + h*64 + col]
      o[((size_t)(b * SEQ + q0 + r)) * DIM + h * DK + nt * 16 + nlane] =
          (_Float16)val;
    }
  }
}

// ---------------------------------------------------------------------------
// Kernel 3: out = o @ Wo + bo  (f16 WMMA, f32 out).
// One wave per 16x64 strip (4 accumulators), A-fragment reused by 4 WMMAs.
// ---------------------------------------------------------------------------
__global__ void oproj_kernel(const _Float16* __restrict__ o,    // [4096,1024]
                             const _Float16* __restrict__ WoT,  // [1024,1024]
                             const float* __restrict__ bo,
                             float* __restrict__ out) {
  const int lane  = threadIdx.x & 31;
  const int wave  = threadIdx.x >> 5;
  const int m0    = blockIdx.x * 16;
  const int strip = blockIdx.y * 4 + wave;    // 0..15
  const int n0    = strip * 64;

  v8f acc[4] = {};
#pragma unroll 2
  for (int k = 0; k < DIM; k += 32) {
    v16h a = frag_f16(o + (size_t)m0 * DIM + k, DIM, lane);
#pragma unroll
    for (int j = 0; j < 4; ++j) {
      v16h b = frag_f16(WoT + (n0 + 16 * j) * DIM + k, DIM, lane);
      acc[j] = wmma_f16(a, b, acc[j]);
    }
  }
  const int mb = (lane & 16) ? 8 : 0;
#pragma unroll
  for (int j = 0; j < 4; ++j) {
    const int n  = n0 + 16 * j + (lane & 15);
    const float bs = bo[n];
#pragma unroll
    for (int i = 0; i < 8; ++i)
      out[(size_t)(m0 + mb + i) * DIM + n] = acc[j][i] + bs;
  }
}

// ---------------------------------------------------------------------------
extern "C" void kernel_launch(void* const* d_in, const int* in_sizes, int n_in,
                              void* d_out, int out_size, void* d_ws, size_t ws_size,
                              hipStream_t stream) {
  (void)in_sizes; (void)n_in; (void)out_size; (void)ws_size;

  const float* Q    = (const float*)d_in[0];
  const float* prev = (const float*)d_in[1];
  const float* Wq   = (const float*)d_in[2];
  const float* bq   = (const float*)d_in[3];
  const float* Wk   = (const float*)d_in[4];
  const float* bk   = (const float*)d_in[5];
  const float* Wv   = (const float*)d_in[6];
  const float* bv   = (const float*)d_in[7];
  const float* Wo   = (const float*)d_in[8];
  const float* bo   = (const float*)d_in[9];

  float* out        = (float*)d_out;                       // [2,2048,1024]
  float* scores_out = out + (size_t)BSZ * SEQ * DIM;       // [2,16,2048,2048]

  // Workspace carve-up (f16), 40 MB total.
  _Float16* ws   = (_Float16*)d_ws;
  const size_t W1M = (size_t)DIM * DIM;       // 1,048,576
  const size_t A4M = (size_t)BSZ * SEQ * DIM; // 4,194,304
  _Float16* WqT  = ws;
  _Float16* WkT  = WqT + W1M;
  _Float16* WvT  = WkT + W1M;
  _Float16* WoT  = WvT + W1M;
  _Float16* qf16 = WoT + W1M;
  _Float16* kf16 = qf16 + A4M;
  _Float16* vT   = kf16 + A4M;
  _Float16* oact = vT + A4M;

  // 0) weight transpose+convert
  wtrans_kernel<<<dim3(4096, 4), 256, 0, stream>>>(Wq, Wk, Wv, Wo,
                                                   WqT, WkT, WvT, WoT);
  // 1) QKV projection + RoPE (z selects q/k/v); 16x64 strips per wave
  qkv_rope_kernel<<<dim3(256, 4, 3), 128, 0, stream>>>(
      Q, WqT, WkT, WvT, bq, bk, bv, qf16, kf16, vT);

  // 2) attention core: 16-row q tiles, 136 KB dynamic LDS
  const int lds_bytes = (16 * SEQ + 8 * 256) * (int)sizeof(float);
  (void)hipFuncSetAttribute((const void*)attn_kernel,
                            hipFuncAttributeMaxDynamicSharedMemorySize,
                            lds_bytes);
  attn_kernel<<<dim3(SEQ / 16, NH, BSZ), 256, lds_bytes, stream>>>(
      qf16, kf16, vT, prev, scores_out, oact);

  // 3) output projection; 16x64 strips per wave
  oproj_kernel<<<dim3(256, 4), 128, 0, stream>>>(oact, WoT, bo, out);
}
